// ST_sampling_1898375545408
// MI455X (gfx1250) — matrix operand order repaired and checked by reading, not compiled, not run on hardware
//
#include <hip/hip_runtime.h>

typedef __attribute__((ext_vector_type(16))) _Float16 v16h;
typedef __attribute__((ext_vector_type(8)))  _Float16 v8h;
typedef __attribute__((ext_vector_type(8)))  float    v8f;

#define DT      300      // crop length
#define KP      320      // K padded to multiple of 32
#define NCROPS  64       // crops per (b,c)
#define FBAND   35       // band bins 7..41
#define NB      80       // N padded (70 used: 35 cos + 35 sin)
#define TLEN    9000
#define KSTEPS  (KP/32)  // 10
#define NTILES  (NB/16)  // 5
#define MTILES  (NCROPS/16) // 4
#define BSWZ_HALVES (NTILES*KSTEPS*32*16)  // 25600 f16 = 51.2 KB
#define BSWZ_CHUNKS (BSWZ_HALVES/8)        // 3200 x 16B

// ---------------------------------------------------------------------------
// Init kernel: build the DFT basis, pre-swizzled into WMMA B-fragment order.
// B fragment (32x16 f16, contiguous K chosen here, as A): lane l -> col N = l&15, K = p + 16*(l>>4), p=0..15.
// Flat address: (((ntile*KSTEPS + s)*32 + lane)*16 + p), so each lane's 16
// halves for a given (ntile, s) are 32 contiguous, 32B-aligned bytes.
// Columns 0..34 = cos(2*pi*f*t/300), 35..69 = sin(...), f = bin 7..41.
// t >= 300 or n >= 70 -> 0 (exact zero-padding of the GEMM).
// ---------------------------------------------------------------------------
__global__ void init_basis_kernel(_Float16* __restrict__ bswz) {
  int idx = blockIdx.x * blockDim.x + threadIdx.x;
  if (idx >= BSWZ_HALVES) return;
  int p     = idx & 15;
  int lane  = (idx >> 4) & 31;
  int g     = idx >> 9;           // ntile*KSTEPS + s
  int s     = g % KSTEPS;
  int ntile = g / KSTEPS;
  int h     = lane >> 4;
  int n     = ntile * 16 + (lane & 15);
  int t     = 32 * s + 16 * h + p;
  float val = 0.0f;
  if (t < DT && n < 2 * FBAND) {
    int f = (n < FBAND ? n : n - FBAND) + 7;     // band bins 7..41
    float ang = 6.283185307179586f * (float)f * (float)t / (float)DT;
    val = (n < FBAND) ? cosf(ang) : sinf(ang);   // sin sign irrelevant (squared)
  }
  bswz[idx] = (_Float16)val;
}

// ---------------------------------------------------------------------------
// Main kernel: one block per (b,c). 256 threads = 8 waves.
//   Phase 0: kick off async copy of the swizzled basis into LDS
//            (global_load_async_to_lds_b128, tracked by ASYNCcnt).
//   Phase 1: gather 64 crops (f32 -> f16) into dense LDS A-panel [64][320]
//            while the async copy is in flight.
//   Phase 2: Y[64][80] = A[64][320] x W[320][80] via v_wmma_f32_16x16x32_f16.
//            Wave w owns M-tile (w&3); A fragments loaded once, reused
//            across its N-tiles. Scalar (SGPR) loop control everywhere.
//   Phase 3: power = cos^2 + sin^2 per bin, normalize over 35 bins, store.
// Mean removal / forward-norm are provably no-ops for the banded, normalized
// output and are skipped.
// ---------------------------------------------------------------------------
__launch_bounds__(256)
__global__ void st_sampling_kernel(const float* __restrict__ in,
                                   const int* __restrict__ offs,
                                   const _Float16* __restrict__ bswz,
                                   float* __restrict__ out) {
  __shared__ __align__(16) _Float16 Apanel[NCROPS][KP];   // 40 KB
  __shared__ __align__(32) _Float16 Bsh[BSWZ_HALVES];     // 51.2 KB
  __shared__ float Ysh[NCROPS][NB];                       // 20.5 KB
  __shared__ int offsh[NCROPS];

  const int tid = threadIdx.x;
  const int bc  = blockIdx.x;                // 0..B*C-1
  const float* chan = in + (size_t)bc * TLEN;

  if (tid < NCROPS) offsh[tid] = offs[bc * NCROPS + tid];

  // ---- Phase 0: async bulk copy of basis blob into LDS (16B per lane). ----
  // Generic shared-pointer low 32 bits == LDS byte address (ISA aperture map).
  {
    const uint32_t lds_base = (uint32_t)(uintptr_t)&Bsh[0];
    for (int i = tid; i < BSWZ_CHUNKS; i += 256) {
      const uint32_t ldsa = lds_base + (uint32_t)i * 16u;
      const uint32_t goff = (uint32_t)i * 16u;
      asm volatile("global_load_async_to_lds_b128 %0, %1, %2"
                   :
                   : "v"(ldsa), "v"(goff), "s"(bswz)
                   : "memory");
    }
  }
  __syncthreads();   // offsh visible; async copy still in flight

  // ---- Phase 1: dense crop panel, f32 -> f16, zero-pad t in [300,320). ----
  for (int e = tid; e < NCROPS * KP; e += 256) {
    int crop = e / KP;
    int t    = e - crop * KP;
    float v  = (t < DT) ? chan[offsh[crop] + t] : 0.0f;
    Apanel[crop][t] = (_Float16)v;
  }
  asm volatile("s_wait_asynccnt 0" ::: "memory");  // basis copy complete
  __syncthreads();

  // ---- Phase 2: WMMA GEMM with scalar loop control. --------------------
  const int lane = tid & 31;
  const int wave = __builtin_amdgcn_readfirstlane(tid) >> 5;  // SGPR
  const int h    = lane >> 4;     // half-wave select
  const int r15  = lane & 15;
  const int m    = wave & 3;      // this wave's M-tile (fixed)
  const int n0   = wave >> 2;     // N-tile start: waves 0-3 -> {0,2,4}, 4-7 -> {1,3}
  const int row  = m * 16 + r15;  // A layout: lane -> matrix row

  // Load this wave's 10 A fragments once (contiguous K order chosen here, not the ISA table order:
  // lane holds K = {h*16+0..15} as two aligned 16B runs, matching B).
  union AF { v16h v; v8h q[2]; };
  AF a[KSTEPS];
#pragma unroll
  for (int s = 0; s < KSTEPS; ++s) {
    a[s].q[0] = *(const v8h*)&Apanel[row][32 * s + 16 * h];
    a[s].q[1] = *(const v8h*)&Apanel[row][32 * s + 16 * h + 8];
  }

  for (int ntile = n0; ntile < NTILES; ntile += 2) {
    v8f acc = {};
#pragma unroll
    for (int s = 0; s < KSTEPS; ++s) {
      // Pre-swizzled B: one contiguous, 32B-aligned 16-half run per lane.
      v16h b = *(const v16h*)&Bsh[(((ntile * KSTEPS + s) * 32) + lane) * 16];
      acc = __builtin_amdgcn_wmma_f32_16x16x32_f16(
          /*neg_a=*/false, a[s].v, /*neg_b=*/false, b,
          /*c_mod=*/(short)0, acc, /*reuse_a=*/false, /*reuse_b=*/false);
    }
    // C/D 16x16 f32 layout: lane -> col N = lane&15, VGPR v -> row v + 8*h.
    const int col = ntile * 16 + r15;
    const int rb  = m * 16 + h * 8;
#pragma unroll
    for (int v = 0; v < 8; ++v) Ysh[rb + v][col] = acc[v];
  }
  __syncthreads();

  // ---- Phase 3: power spectrum in band + unit-sum normalization. -------
  if (tid < NCROPS) {
    float p[FBAND];
    float ssum = 0.0f;
#pragma unroll
    for (int f = 0; f < FBAND; ++f) {
      float re = Ysh[tid][f];
      float im = Ysh[tid][FBAND + f];
      p[f] = re * re + im * im;
      ssum += p[f];
    }
    float inv = 1.0f / ssum;
    float* o = out + ((size_t)bc * NCROPS + tid) * FBAND;
#pragma unroll
    for (int f = 0; f < FBAND; ++f) o[f] = p[f] * inv;
  }
}

extern "C" void kernel_launch(void* const* d_in, const int* in_sizes, int n_in,
                              void* d_out, int out_size, void* d_ws, size_t ws_size,
                              hipStream_t stream) {
  const float* in  = (const float*)d_in[0];   // [256,16,9000] f32
  const int* offs  = (const int*)d_in[1];     // [256,16,64] i32
  float* out       = (float*)d_out;           // [256,16,64,35] f32
  _Float16* bswz   = (_Float16*)d_ws;         // needs 51,200 bytes of scratch

  init_basis_kernel<<<(BSWZ_HALVES + 255) / 256, 256, 0, stream>>>(bswz);
  st_sampling_kernel<<<256 * 16, 256, 0, stream>>>(in, offs, bswz, out);
}
